// Decoder_9234179686487
// MI455X (gfx1250) — matrix operand order repaired and checked by reading, not compiled, not run on hardware
//
#include <hip/hip_runtime.h>
#include <math.h>

// ---------------- CDNA5 WMMA types ----------------
typedef __attribute__((ext_vector_type(16))) __bf16    v16bf;
typedef __attribute__((ext_vector_type(8)))  float     v8f;
typedef __attribute__((ext_vector_type(8)))  unsigned  v8u;
typedef unsigned short u16;

// gfx1250 async global->LDS path (guarded: fall back to sync staging if the
// toolchain doesn't declare the builtins). Probe round 2 showed the builtin's
// params are (int4 addrspace(1)*, int4 addrspace(3)*, imm int, imm int).
#if __has_builtin(__builtin_amdgcn_global_load_async_to_lds_b128) && \
    __has_builtin(__builtin_amdgcn_s_wait_asynccnt)
#define ASYNC_LDS 1
typedef int v4i __attribute__((vector_size(16)));
typedef __attribute__((address_space(1))) v4i as1_v4i;
typedef __attribute__((address_space(3))) v4i as3_v4i;
#else
#define ASYNC_LDS 0
#endif

__device__ __forceinline__ v8f wmma_bf16(v16bf a, v16bf b, v8f c) {
  // D = A(16x32 bf16) * B(32x16 bf16) + C(16x16 f32)
  return __builtin_amdgcn_wmma_f32_16x16x32_bf16(false, a, false, b, (short)0, c,
                                                 false, false);
}

__device__ __forceinline__ u16 f2bf(float f) {
  unsigned u = __builtin_bit_cast(unsigned, f);
  unsigned r = u + 0x7FFFu + ((u >> 16) & 1u);   // round-to-nearest-even
  return (u16)(r >> 16);
}

// A-fragment (16x32, MxK): lane(0-15)=row M, K = h*16 + vg*2 + p
// src is row-major [16 rows][K cols], row stride = `stride` elements.
__device__ __forceinline__ v16bf load_a(const u16* base, int stride, int lane) {
  const int h = lane >> 4, r = lane & 15;
  v8u f;
#pragma unroll
  for (int vg = 0; vg < 8; ++vg) {
    const int k0 = h * 16 + vg * 2;
    f[vg] = *(const unsigned*)(base + (size_t)r * stride + k0);
  }
  return __builtin_bit_cast(v16bf, f);
}

// B-fragment (32x16, KxN): lane(0-15)=col N, K = h*16 + vg*2 + p
// src is "B-transposed" row-major [16 N-rows][K cols], row stride = `stride`.
__device__ __forceinline__ v16bf load_b(const u16* base, int stride, int lane) {
  const int h = lane >> 4, n = lane & 15;
  v8u f;
#pragma unroll
  for (int vg = 0; vg < 8; ++vg) {
    const int k0 = h * 16 + vg * 2;
    f[vg] = *(const unsigned*)(base + (size_t)n * stride + k0);
  }
  return __builtin_bit_cast(v16bf, f);
}

// ---------------- small elementwise kernels ----------------

__global__ void pe_kernel(float* pe, int L, int C) {
  int idx = blockIdx.x * blockDim.x + threadIdx.x;
  if (idx >= L * C) return;
  int l = idx / C, c = idx % C;
  float e = (float)(c & ~1) / (float)C;
  float item = __powf(10000.0f, e);
  float v = (float)l / item;
  pe[idx] = (c & 1) ? __cosf(v) : __sinf(v);
}

// Wt[n*K + k] = bf16(W[k*N + n])   (transpose + convert weights)
__global__ void wt_cvt_kernel(const float* W, u16* Wt, int K, int N) {
  int idx = blockIdx.x * blockDim.x + threadIdx.x;
  if (idx >= K * N) return;
  int n = idx / K, k = idx - n * K;
  Wt[(size_t)n * K + k] = f2bf(W[(size_t)k * N + n]);
}

__global__ void addpe_cvt_kernel(const float* x, const float* pe, u16* out,
                                 long total, int LC) {
  long idx = (long)blockIdx.x * blockDim.x + threadIdx.x;
  if (idx >= total) return;
  out[idx] = f2bf(x[idx] + pe[idx % LC]);
}

__global__ void cvt_kernel(const float* x, u16* out, long total) {
  long idx = (long)blockIdx.x * blockDim.x + threadIdx.x;
  if (idx >= total) return;
  out[idx] = f2bf(x[idx]);
}

// Vout[((b*H+h)*D + d)*LK + k] = Vin[(b*LK + k)*C + h*D + d]
__global__ void vtrans_kernel(const u16* Vin, u16* Vout, int Bn, int LK) {
  const int C = 512, D = 64, H = 8;
  long idx = (long)blockIdx.x * blockDim.x + threadIdx.x;
  long total = (long)Bn * LK * C;
  if (idx >= total) return;
  int k = idx % LK;
  int d = (idx / LK) % D;
  int h = (idx / ((long)LK * D)) % H;
  int b = idx / ((long)LK * D * H);
  Vout[idx] = Vin[((size_t)(b * LK + k)) * C + h * D + d];
}

// out = LN(x + res) * g + b ; optional bf16 copy
__global__ __launch_bounds__(256) void ln_kernel(const float* x, const float* res,
                                                 const float* g, const float* beta,
                                                 float* outf, u16* outb, int C) {
  const int row = blockIdx.x, tid = threadIdx.x;
  __shared__ float ssum[256], ssq[256];
  const float* xr = x + (size_t)row * C;
  const float* rr = res + (size_t)row * C;
  float s = 0.f, q = 0.f;
  for (int c = tid; c < C; c += 256) {
    float v = xr[c] + rr[c];
    s += v; q += v * v;
  }
  ssum[tid] = s; ssq[tid] = q;
  __syncthreads();
  for (int off = 128; off > 0; off >>= 1) {
    if (tid < off) { ssum[tid] += ssum[tid + off]; ssq[tid] += ssq[tid + off]; }
    __syncthreads();
  }
  float mean = ssum[0] / (float)C;
  float var = ssq[0] / (float)C - mean * mean;
  float inv = rsqrtf(var + 1e-5f);
  for (int c = tid; c < C; c += 256) {
    float v = (xr[c] + rr[c] - mean) * inv * g[c] + beta[c];
    outf[(size_t)row * C + c] = v;
    if (outb) outb[(size_t)row * C + c] = f2bf(v);
  }
}

// ---------------- WMMA GEMM: Y[M,N] = A[M,K](bf16) @ W + bias ----------------
// W pre-transposed as Wt[N,K] bf16. Block = 4 waves, each wave one 16x64 strip
// over a shared 64-wide N range: the 64x32 Wt tile is staged once per block in
// LDS (double-buffered async global->LDS copies on CDNA5), then all 4 waves
// read B-fragments from LDS while A-fragments stream from global.
template <bool RELU, bool OUTBF>
__global__ __launch_bounds__(128) void gemm_kernel(const u16* __restrict__ A,
                                                   const u16* __restrict__ Wt,
                                                   const float* __restrict__ bias,
                                                   float* Yf, u16* Yb,
                                                   int M, int N, int K) {
  __shared__ u16 bs[2][64 * 32];   // 2 x 4KB B tiles
  const int tid = threadIdx.x;
  const int lane = tid & 31;
  const int wv = tid >> 5;
  const int m0 = (blockIdx.x * 4 + wv) * 16;
  const int n0 = blockIdx.y * 64;
  const int hl = lane >> 4, ln = lane & 15;

  // staging map: thread t copies 32B of Wt row (t>>1), halfword col (t&1)*16
  const int srow = tid >> 1;
  const int scol = (tid & 1) * 16;

  v8f acc[4] = {v8f{0}, v8f{0}, v8f{0}, v8f{0}};
  const u16* Arow = A + (size_t)m0 * K;
  const int nk = K / 32;

#if ASYNC_LDS
#define GEMM_STAGE(buf, kkk)                                                   \
  {                                                                            \
    const u16* gsrc = Wt + (size_t)(n0 + srow) * K + (kkk) + scol;             \
    u16* ldst = &bs[buf][srow * 32 + scol];                                    \
    __builtin_amdgcn_global_load_async_to_lds_b128((as1_v4i*)gsrc,             \
                                                   (as3_v4i*)ldst, 0, 0);      \
    __builtin_amdgcn_global_load_async_to_lds_b128((as1_v4i*)(gsrc + 8),       \
                                                   (as3_v4i*)(ldst + 8), 0, 0);\
  }
  GEMM_STAGE(0, 0);
  for (int i = 0; i < nk; ++i) {
    const int kk = i * 32;
    if (i + 1 < nk) {
      GEMM_STAGE((i + 1) & 1, kk + 32);
      __builtin_amdgcn_s_wait_asynccnt(2);  // current tile's copies done
    } else {
      __builtin_amdgcn_s_wait_asynccnt(0);
    }
    __syncthreads();                        // staged data visible to all waves
    __builtin_prefetch(Arow + kk + 128, 0, 1);
    v16bf a = load_a(Arow + kk, K, lane);
    const u16* bbase = &bs[i & 1][0];
#pragma unroll
    for (int j = 0; j < 4; ++j)
      acc[j] = wmma_bf16(a, load_b(bbase + j * 16 * 32, 32, lane), acc[j]);
    __syncthreads();                        // protect buffer before reuse
  }
#undef GEMM_STAGE
#else
  for (int i = 0; i < nk; ++i) {
    const int kk = i * 32;
    __syncthreads();
    {
      const u16* gsrc = Wt + (size_t)(n0 + srow) * K + kk + scol;
      *(v8u*)&bs[0][srow * 32 + scol] = *(const v8u*)gsrc;  // 32B sync copy
    }
    __syncthreads();
    __builtin_prefetch(Arow + kk + 128, 0, 1);
    v16bf a = load_a(Arow + kk, K, lane);
#pragma unroll
    for (int j = 0; j < 4; ++j)
      acc[j] = wmma_bf16(a, load_b(&bs[0][j * 16 * 32], 32, lane), acc[j]);
  }
#endif

#pragma unroll
  for (int j = 0; j < 4; ++j) {
    const int n = n0 + j * 16 + ln;
    const float bv = bias[n];
#pragma unroll
    for (int v = 0; v < 8; ++v) {
      const int m = m0 + v + 8 * hl;
      float val = acc[j][v] + bv;
      if (RELU) val = fmaxf(val, 0.f);
      if (OUTBF) Yb[(size_t)m * N + n] = f2bf(val);
      else       Yf[(size_t)m * N + n] = val;
    }
  }
}

// ---------------- Flash attention (WMMA, online softmax) ----------------
// Q: [B*LQ, C] bf16 ; K: [B*LK, C] bf16 ; Vt: [B,H,D,LK] bf16 ; O: [B*LQ, C] bf16
#define ATT_WAVES 4
__global__ __launch_bounds__(32 * ATT_WAVES) void attn_kernel(
    const u16* __restrict__ Q, const u16* __restrict__ K,
    const u16* __restrict__ Vt, u16* __restrict__ O,
    int LQ, int LK, float scale) {
  const int C = 512, D = 64, H = 8;
  const int lane = threadIdx.x & 31;
  const int wv = threadIdx.x >> 5;
  const int q0 = (blockIdx.x * ATT_WAVES + wv) * 16;
  const int h = blockIdx.y;
  const int b = blockIdx.z;
  const int hl = lane >> 4, ln = lane & 15;

  __shared__ u16 plds[ATT_WAVES][16 * 32];
  u16* pl = plds[wv];

  const u16* Qbase = Q + ((size_t)(b * LQ + q0)) * C + h * D;
  const u16* Kbase = K + ((size_t)b * LK) * C + h * D;
  const u16* Vbase = Vt + ((size_t)(b * H + h) * D) * (size_t)LK;

  const v16bf qa0 = load_a(Qbase + 0, C, lane);
  const v16bf qa1 = load_a(Qbase + 32, C, lane);

  v8f acc[4] = {v8f{0}, v8f{0}, v8f{0}, v8f{0}};
  float mrow[8], lrow[8];
#pragma unroll
  for (int v = 0; v < 8; ++v) { mrow[v] = -1e30f; lrow[v] = 0.f; }

  const int nkt = LK / 32;
  for (int kt = 0; kt < nkt; ++kt) {
    // ---- scores: S (16 q-rows x 32 keys) as two 16x16 C-tiles ----
    const u16* kb0 = Kbase + (size_t)(kt * 32) * C;
    const u16* kb1 = kb0 + (size_t)16 * C;
    v8f s0 = v8f{0}, s1 = v8f{0};
    s0 = wmma_bf16(qa0, load_b(kb0 + 0,  C, lane), s0);
    s0 = wmma_bf16(qa1, load_b(kb0 + 32, C, lane), s0);
    s1 = wmma_bf16(qa0, load_b(kb1 + 0,  C, lane), s1);
    s1 = wmma_bf16(qa1, load_b(kb1 + 32, C, lane), s1);

    // ---- online softmax (row stats reduced across the 16-lane half) ----
    float p0[8], p1[8];
#pragma unroll
    for (int v = 0; v < 8; ++v) {
      float a0 = s0[v] * scale, a1 = s1[v] * scale;
      float tm = fmaxf(a0, a1);
#pragma unroll
      for (int msk = 1; msk < 16; msk <<= 1) tm = fmaxf(tm, __shfl_xor(tm, msk, 32));
      float mnew = fmaxf(mrow[v], tm);
      float alpha = __expf(mrow[v] - mnew);
      p0[v] = __expf(a0 - mnew);
      p1[v] = __expf(a1 - mnew);
      float rs = p0[v] + p1[v];
#pragma unroll
      for (int msk = 1; msk < 16; msk <<= 1) rs += __shfl_xor(rs, msk, 32);
      lrow[v] = lrow[v] * alpha + rs;
      mrow[v] = mnew;
      acc[0][v] *= alpha; acc[1][v] *= alpha; acc[2][v] *= alpha; acc[3][v] *= alpha;
    }

    // ---- reshape P: C-layout -> A-layout via LDS ----
    __syncthreads();
#pragma unroll
    for (int v = 0; v < 8; ++v) {
      pl[(v + 8 * hl) * 32 + ln]      = f2bf(p0[v]);
      pl[(v + 8 * hl) * 32 + 16 + ln] = f2bf(p1[v]);
    }
    __syncthreads();
    v16bf pa = load_a(pl, 32, lane);

    // ---- acc += P(16x32) @ V(32x64) ----
    const u16* vb = Vbase + kt * 32;
#pragma unroll
    for (int j = 0; j < 4; ++j)
      acc[j] = wmma_bf16(pa, load_b(vb + (size_t)(j * 16) * LK, LK, lane), acc[j]);
  }

  // ---- normalize and write O[b, q, h*D + d] as bf16 ----
#pragma unroll
  for (int j = 0; j < 4; ++j) {
#pragma unroll
    for (int v = 0; v < 8; ++v) {
      const int m = q0 + v + 8 * hl;
      const int col = h * D + j * 16 + ln;
      O[((size_t)(b * LQ + m)) * C + col] = f2bf(acc[j][v] / lrow[v]);
    }
  }
}

// ---------------- host orchestration ----------------
static inline void* wsa(char*& p, size_t bytes) {
  void* r = (void*)p;
  p += (bytes + 255) & ~(size_t)255;
  return r;
}

extern "C" void kernel_launch(void* const* d_in, const int* in_sizes, int n_in,
                              void* d_out, int out_size, void* d_ws, size_t ws_size,
                              hipStream_t stream) {
  (void)in_sizes; (void)n_in; (void)out_size; (void)ws_size;
  const int B = 16, LQ = 256, LM = 4096, C = 512, F = 2048;
  const int MQ = B * LQ;       // 4096
  const int MM = B * LM;       // 65536
  const float scale = 0.125f;  // 1/sqrt(64)

  const float* query  = (const float*)d_in[0];
  const float* memory = (const float*)d_in[1];
  const float* sa_wq = (const float*)d_in[2];  const float* sa_bq = (const float*)d_in[3];
  const float* sa_wk = (const float*)d_in[4];  const float* sa_bk = (const float*)d_in[5];
  const float* sa_wv = (const float*)d_in[6];  const float* sa_bv = (const float*)d_in[7];
  const float* sa_wo = (const float*)d_in[8];  const float* sa_bo = (const float*)d_in[9];
  const float* ca_wq = (const float*)d_in[10]; const float* ca_bq = (const float*)d_in[11];
  const float* ca_wk = (const float*)d_in[12]; const float* ca_bk = (const float*)d_in[13];
  const float* ca_wv = (const float*)d_in[14]; const float* ca_bv = (const float*)d_in[15];
  const float* ca_wo = (const float*)d_in[16]; const float* ca_bo = (const float*)d_in[17];
  const float* ffn_w1 = (const float*)d_in[18]; const float* ffn_b1 = (const float*)d_in[19];
  const float* ffn_w2 = (const float*)d_in[20]; const float* ffn_b2 = (const float*)d_in[21];
  const float* ln1_g = (const float*)d_in[22]; const float* ln1_b = (const float*)d_in[23];
  const float* ln2_g = (const float*)d_in[24]; const float* ln2_b = (const float*)d_in[25];
  const float* ln3_g = (const float*)d_in[26]; const float* ln3_b = (const float*)d_in[27];
  float* out = (float*)d_out;

  // ---- workspace layout ----
  char* p = (char*)d_ws;
  u16* wt_saq = (u16*)wsa(p, (size_t)C * C * 2);
  u16* wt_sak = (u16*)wsa(p, (size_t)C * C * 2);
  u16* wt_sav = (u16*)wsa(p, (size_t)C * C * 2);
  u16* wt_sao = (u16*)wsa(p, (size_t)C * C * 2);
  u16* wt_caq = (u16*)wsa(p, (size_t)C * C * 2);
  u16* wt_cak = (u16*)wsa(p, (size_t)C * C * 2);
  u16* wt_cav = (u16*)wsa(p, (size_t)C * C * 2);
  u16* wt_cao = (u16*)wsa(p, (size_t)C * C * 2);
  u16* wt_f1  = (u16*)wsa(p, (size_t)C * F * 2);
  u16* wt_f2  = (u16*)wsa(p, (size_t)F * C * 2);
  float* pe_q = (float*)wsa(p, (size_t)LQ * C * 4);
  float* pe_m = (float*)wsa(p, (size_t)LM * C * 4);
  u16* xq_pe  = (u16*)wsa(p, (size_t)MQ * C * 2);   // reused as qc_in later
  u16* xq_bf  = (u16*)wsa(p, (size_t)MQ * C * 2);
  u16* Qb     = (u16*)wsa(p, (size_t)MQ * C * 2);   // Q proj (self, then cross)
  u16* Kb     = (u16*)wsa(p, (size_t)MQ * C * 2);   // self K proj
  u16* Vtmp   = (u16*)wsa(p, (size_t)MQ * C * 2);   // self V proj (pre-transpose)
  u16* Vst    = (u16*)wsa(p, (size_t)MQ * C * 2);   // self V transposed
  u16* Oatt   = (u16*)wsa(p, (size_t)MQ * C * 2);   // attn out (self, then cross)
  float* projf = (float*)wsa(p, (size_t)MQ * C * 4); // out-proj f32 / ffn2 f32
  float* tgt   = (float*)wsa(p, (size_t)MQ * C * 4);
  float* tgt2  = (float*)wsa(p, (size_t)MQ * C * 4);
  u16* tgt2b  = (u16*)wsa(p, (size_t)MQ * C * 2);
  u16* h1     = (u16*)wsa(p, (size_t)MQ * F * 2);
  u16* m_pe   = (u16*)wsa(p, (size_t)MM * C * 2);   // reused as Vct after Kc
  u16* m_cv   = (u16*)wsa(p, (size_t)MM * C * 2);
  u16* Kc     = (u16*)wsa(p, (size_t)MM * C * 2);
  u16* Vc_tmp = (u16*)wsa(p, (size_t)MM * C * 2);
  u16* Vct    = m_pe;                                // alias: m_pe dead after Kc

  const int T = 256;
  auto blks = [&](long n) { return (unsigned)((n + T - 1) / T); };

  // ---- weight transpose+convert, PE tables ----
  wt_cvt_kernel<<<blks((long)C * C), T, 0, stream>>>(sa_wq, wt_saq, C, C);
  wt_cvt_kernel<<<blks((long)C * C), T, 0, stream>>>(sa_wk, wt_sak, C, C);
  wt_cvt_kernel<<<blks((long)C * C), T, 0, stream>>>(sa_wv, wt_sav, C, C);
  wt_cvt_kernel<<<blks((long)C * C), T, 0, stream>>>(sa_wo, wt_sao, C, C);
  wt_cvt_kernel<<<blks((long)C * C), T, 0, stream>>>(ca_wq, wt_caq, C, C);
  wt_cvt_kernel<<<blks((long)C * C), T, 0, stream>>>(ca_wk, wt_cak, C, C);
  wt_cvt_kernel<<<blks((long)C * C), T, 0, stream>>>(ca_wv, wt_cav, C, C);
  wt_cvt_kernel<<<blks((long)C * C), T, 0, stream>>>(ca_wo, wt_cao, C, C);
  wt_cvt_kernel<<<blks((long)C * F), T, 0, stream>>>(ffn_w1, wt_f1, C, F);
  wt_cvt_kernel<<<blks((long)F * C), T, 0, stream>>>(ffn_w2, wt_f2, F, C);
  pe_kernel<<<blks((long)LQ * C), T, 0, stream>>>(pe_q, LQ, C);
  pe_kernel<<<blks((long)LM * C), T, 0, stream>>>(pe_m, LM, C);

  // ---- self attention ----
  addpe_cvt_kernel<<<blks((long)MQ * C), T, 0, stream>>>(query, pe_q, xq_pe,
                                                         (long)MQ * C, LQ * C);
  cvt_kernel<<<blks((long)MQ * C), T, 0, stream>>>(query, xq_bf, (long)MQ * C);

  dim3 gq(MQ / 64, C / 64), gb(128);
  gemm_kernel<false, true><<<gq, gb, 0, stream>>>(xq_pe, wt_saq, sa_bq, nullptr, Qb,   MQ, C, C);
  gemm_kernel<false, true><<<gq, gb, 0, stream>>>(xq_pe, wt_sak, sa_bk, nullptr, Kb,   MQ, C, C);
  gemm_kernel<false, true><<<gq, gb, 0, stream>>>(xq_bf, wt_sav, sa_bv, nullptr, Vtmp, MQ, C, C);
  vtrans_kernel<<<blks((long)MQ * C), T, 0, stream>>>(Vtmp, Vst, B, LQ);

  dim3 ga(LQ / (16 * ATT_WAVES), 8, B);
  attn_kernel<<<ga, 32 * ATT_WAVES, 0, stream>>>(Qb, Kb, Vst, Oatt, LQ, LQ, scale);

  gemm_kernel<false, false><<<gq, gb, 0, stream>>>(Oatt, wt_sao, sa_bo, projf, nullptr, MQ, C, C);
  ln_kernel<<<MQ, 256, 0, stream>>>(projf, query, ln1_g, ln1_b, tgt, nullptr, C);

  // ---- cross attention ----
  addpe_cvt_kernel<<<blks((long)MQ * C), T, 0, stream>>>(tgt, pe_q, xq_pe,
                                                         (long)MQ * C, LQ * C);
  addpe_cvt_kernel<<<blks((long)MM * C), T, 0, stream>>>(memory, pe_m, m_pe,
                                                         (long)MM * C, LM * C);
  cvt_kernel<<<blks((long)MM * C), T, 0, stream>>>(memory, m_cv, (long)MM * C);

  dim3 gm(MM / 64, C / 64);
  gemm_kernel<false, true><<<gq, gb, 0, stream>>>(xq_pe, wt_caq, ca_bq, nullptr, Qb, MQ, C, C);
  gemm_kernel<false, true><<<gm, gb, 0, stream>>>(m_pe, wt_cak, ca_bk, nullptr, Kc, MM, C, C);
  gemm_kernel<false, true><<<gm, gb, 0, stream>>>(m_cv, wt_cav, ca_bv, nullptr, Vc_tmp, MM, C, C);
  vtrans_kernel<<<blks((long)MM * C), T, 0, stream>>>(Vc_tmp, Vct, B, LM);

  attn_kernel<<<ga, 32 * ATT_WAVES, 0, stream>>>(Qb, Kc, Vct, Oatt, LQ, LM, scale);

  gemm_kernel<false, false><<<gq, gb, 0, stream>>>(Oatt, wt_cao, ca_bo, projf, nullptr, MQ, C, C);
  ln_kernel<<<MQ, 256, 0, stream>>>(projf, tgt, ln2_g, ln2_b, tgt2, tgt2b, C);

  // ---- FFN ----
  dim3 gf1(MQ / 64, F / 64);
  gemm_kernel<true, true><<<gf1, gb, 0, stream>>>(tgt2b, wt_f1, ffn_b1, nullptr, h1, MQ, F, C);
  gemm_kernel<false, false><<<gq, gb, 0, stream>>>(h1, wt_f2, ffn_b2, projf, nullptr, MQ, C, F);
  ln_kernel<<<MQ, 256, 0, stream>>>(projf, tgt2, ln3_g, ln3_b, out, nullptr, C);
}
